// CoreModule_82686710382601
// MI455X (gfx1250) — matrix operand order repaired and checked by reading, not compiled, not run on hardware
//
#include <hip/hip_runtime.h>

// ---------------- constants ----------------
#define NN 8192
#define EE 32768
#define DD 512
#define XDIM 64
#define EDIM 16
#define DFF 2048
#define RR 512
#define RSZ 6
#define BB 64
#define NPER 128
#define RPER 8
#define SEQ 139
#define GATL 6

typedef __attribute__((ext_vector_type(16))) __bf16 v16bf;
typedef __attribute__((ext_vector_type(8)))  float  v8f;

// ---------------- helpers ----------------
__device__ __forceinline__ unsigned int fenc(float f) {
  unsigned int u = __float_as_uint(f);
  return (u & 0x80000000u) ? ~u : (u | 0x80000000u);
}
__device__ __forceinline__ float fdec(unsigned int u) {
  return (u & 0x80000000u) ? __uint_as_float(u ^ 0x80000000u) : __uint_as_float(~u);
}
__device__ __forceinline__ unsigned short f2bf(float f) {
  unsigned int u = __float_as_uint(f);
  return (unsigned short)((u + 0x7FFFu + ((u >> 16) & 1u)) >> 16);
}
__device__ __forceinline__ v16bf frag_from(uint4 lo, uint4 hi) {
  union { uint4 q[2]; v16bf v; } u;
  u.q[0] = lo; u.q[1] = hi;
  return u.v;
}

// ---------------- f32 -> bf16 (raw ushort, RNE), 4 elems/thread ----------------
__global__ void k_cvt_bf16(const float* __restrict__ src, unsigned short* __restrict__ dst, long n) {
  long i = ((long)blockIdx.x * blockDim.x + threadIdx.x) * 4;
  if (i + 4 <= n) {
    float4 f = *(const float4*)(src + i);
    unsigned int lo = (unsigned int)f2bf(f.x) | ((unsigned int)f2bf(f.y) << 16);
    unsigned int hi = (unsigned int)f2bf(f.z) | ((unsigned int)f2bf(f.w) << 16);
    *(uint2*)(dst + i) = make_uint2(lo, hi);
  } else {
    for (; i < n; ++i) dst[i] = f2bf(src[i]);
  }
}

__global__ void k_fill_f32(float* __restrict__ p, float v, long n) {
  long i = (long)blockIdx.x * blockDim.x + threadIdx.x;
  if (i < n) p[i] = v;
}
__global__ void k_fill_u32(unsigned int* __restrict__ p, unsigned int v, long n) {
  long i = (long)blockIdx.x * blockDim.x + threadIdx.x;
  if (i < n) p[i] = v;
}

// ---------------- WMMA bf16 GEMM: C = act(A[MxK] * B[KxN] + bias) ----------------
// Block tile 128x64, BK=32, 256 threads = 8 waves; wave w owns M-tile w (16 rows)
// and all four 16-wide N-tiles -> 4 WMMAs per K-step per wave (issued back-to-back).
// A staged row-major in LDS (fragment = 2x ds_load_b128 per lane).
// B staged TRANSPOSED (column-major) in LDS (fragment = 2x ds_load_b128 per lane).
// Requires: K % 8 == 0 (all call sites: 16, 64, 512, 2048), N % 64 == 0.
#define BM 128
#define BN 64
#define BK 32

template <int ACT>
__global__ __launch_bounds__(256) void k_gemm(const unsigned short* __restrict__ A,
                                              const unsigned short* __restrict__ Bm,
                                              const float* __restrict__ bias,
                                              float* __restrict__ C,
                                              int M, int N, int K) {
  __shared__ unsigned short As[BM][BK];      // row-major, row = 64B (16B-aligned chunks)
  __shared__ unsigned short Bt[BN * BK];     // column-major: Bt[n*32 + k]
  const int tid = threadIdx.x;
  const int wave = tid >> 5, lane = tid & 31;
  const int bm = blockIdx.y * BM, bn = blockIdx.x * BN;
  v8f acc[4] = {};

  for (int k0 = 0; k0 < K; k0 += BK) {
    // ---- stage A: 128x32 = 512 8-elem chunks, 2 per thread (global_load_b128) ----
#pragma unroll
    for (int j = 0; j < 2; ++j) {
      int ci = tid * 2 + j;
      int row = ci >> 2, cc = (ci & 3) * 8;
      int gr = bm + row, gk = k0 + cc;
      uint4 v = {0u, 0u, 0u, 0u};
      if (gr < M && gk + 8 <= K) v = *(const uint4*)&A[(long)gr * K + gk];
      *(uint4*)&As[row][cc] = v;
      if (j == 0 && gr < M && k0 + BK < K) __builtin_prefetch(&A[(long)gr * K + k0 + BK], 0, 3);
    }
    // ---- stage B transposed: 32x64 = 256 chunks, 1 per thread ----
    {
      int krow = tid >> 3, cc = (tid & 7) * 8;
      int gk = k0 + krow, gn = bn + cc;
      uint4 v = {0u, 0u, 0u, 0u};
      if (gk < K && gn + 8 <= N) v = *(const uint4*)&Bm[(long)gk * N + gn];
      const unsigned short* pv = (const unsigned short*)&v;
#pragma unroll
      for (int j = 0; j < 8; ++j) Bt[(cc + j) * BK + krow] = pv[j];
    }
    __syncthreads();

    // A fragment: lanes<16 hold K {0-7,16-23}, lanes>=16 hold K {8-15,24-31}
    const uint4* pa = (const uint4*)&As[wave * 16 + (lane & 15)][0];
    int ka = lane >> 4;  // chunk 0 or 1
    v16bf fa = frag_from(pa[ka], pa[ka + 2]);
    int kb = lane >> 4;  // chunk 0 or 1 (K {0-7,16-23} vs {8-15,24-31})
    // Load ALL B fragments first, then issue WMMAs back-to-back.
    v16bf fb[4];
#pragma unroll
    for (int nt = 0; nt < 4; ++nt) {
      const uint4* pb = (const uint4*)&Bt[(nt * 16 + (lane & 15)) * BK];
      fb[nt] = frag_from(pb[kb], pb[kb + 2]);
    }
#pragma unroll
    for (int nt = 0; nt < 4; ++nt) {
      acc[nt] = __builtin_amdgcn_wmma_f32_16x16x32_bf16(false, fa, false, fb[nt], (short)0, acc[nt], false, false);
    }
    __syncthreads();
  }

  // C/D layout: VGPR r -> row r (+8 for lanes>=16), col = lane&15
  int mbase = bm + wave * 16 + ((lane >> 4) << 3);
#pragma unroll
  for (int nt = 0; nt < 4; ++nt) {
    int nc = bn + nt * 16 + (lane & 15);
    float bv = (bias && nc < N) ? bias[nc] : 0.f;
#pragma unroll
    for (int r = 0; r < 8; ++r) {
      int gm = mbase + r;
      if (gm < M && nc < N) {
        float v = acc[nt][r] + bv;
        if (ACT == 1) v = fmaxf(v, 0.f);
        C[(long)gm * N + nc] = v;
      }
    }
  }
}

// ---------------- GAT edge kernels ----------------
__global__ void k_deg(const int* __restrict__ dst, float* __restrict__ deg) {
  int i = blockIdx.x * blockDim.x + threadIdx.x;
  if (i < EE) atomicAdd(&deg[dst[i]], 1.0f);
}
__global__ void k_loop_sum(const int* __restrict__ dst, float* __restrict__ eall) {
  int e = blockIdx.x;
  int dn = dst[e];
  float* d = eall + ((long)EE + dn) * DD;
  const float* s = eall + (long)e * DD;
  for (int c = threadIdx.x; c < DD; c += blockDim.x) atomicAdd(&d[c], s[c]);
}
__global__ void k_loop_div(const float* __restrict__ deg, float* __restrict__ eall) {
  int n = blockIdx.x;
  float inv = 1.0f / fmaxf(deg[n], 1.0f);
  float* row = eall + ((long)EE + n) * DD;
  for (int c = threadIdx.x; c < DD; c += blockDim.x) row[c] *= inv;
}

__global__ void k_edge_scores(const float* __restrict__ xl, const float* __restrict__ xr,
                              const float* __restrict__ me, const float* __restrict__ att,
                              const int* __restrict__ src, const int* __restrict__ dst,
                              float* __restrict__ s, unsigned int* __restrict__ smax, int ntot) {
  int idx = blockIdx.x * 8 + (threadIdx.x >> 5);
  if (idx >= ntot) return;
  int lane = threadIdx.x & 31;
  int sn, dn;
  if (idx < EE) { sn = src[idx]; dn = dst[idx]; } else { sn = dn = idx - EE; }
  const float* pl = xl + (long)sn * DD;
  const float* pr = xr + (long)dn * DD;
  const float* pm = me + (long)idx * DD;
  float acc = 0.f;
  for (int d = lane; d < DD; d += 32) {
    float m = pl[d] + pr[d] + pm[d];
    m = (m > 0.f) ? m : 0.2f * m;
    acc += m * att[d];
  }
#pragma unroll
  for (int off = 16; off; off >>= 1) acc += __shfl_xor(acc, off, 32);
  if (lane == 0) {
    s[idx] = acc;
    atomicMax(&smax[dn], fenc(acc));
  }
}

__global__ void k_edge_exp(const float* __restrict__ s, const unsigned int* __restrict__ smax,
                           const int* __restrict__ dst, float* __restrict__ ex,
                           float* __restrict__ z, int ntot) {
  int idx = blockIdx.x * blockDim.x + threadIdx.x;
  if (idx >= ntot) return;
  int dn = (idx < EE) ? dst[idx] : idx - EE;
  float e = __expf(s[idx] - fdec(smax[dn]));
  ex[idx] = e;
  atomicAdd(&z[dn], e);
}

__global__ void k_edge_agg(const float* __restrict__ ex, const float* __restrict__ z,
                           const float* __restrict__ xl, const int* __restrict__ src,
                           const int* __restrict__ dst, float* __restrict__ out) {
  int idx = blockIdx.x;
  int sn, dn;
  if (idx < EE) { sn = src[idx]; dn = dst[idx]; } else { sn = dn = idx - EE; }
  float a = ex[idx] / z[dn];
  const float* pl = xl + (long)sn * DD;
  float* po = out + (long)dn * DD;
  for (int d = threadIdx.x; d < DD; d += blockDim.x) atomicAdd(&po[d], a * pl[d]);
}

// ---------------- fused residual+bias+LN(+ReLU), D=512, 256 thr ----------------
__global__ __launch_bounds__(256) void k_row_ln(const float* __restrict__ in,
                                                const float* __restrict__ bias,
                                                const float* __restrict__ resid,
                                                const float* __restrict__ g,
                                                const float* __restrict__ beta,
                                                float* __restrict__ out,
                                                int do_ln, int do_relu) {
  __shared__ float red[256];
  int row = blockIdx.x, t = threadIdx.x;
  long base = (long)row * DD;
  float v0 = in[base + t]       + (bias ? bias[t]       : 0.f) + (resid ? resid[base + t]       : 0.f);
  float v1 = in[base + t + 256] + (bias ? bias[t + 256] : 0.f) + (resid ? resid[base + t + 256] : 0.f);
  if (!do_ln) {
    if (do_relu) { v0 = fmaxf(v0, 0.f); v1 = fmaxf(v1, 0.f); }
    out[base + t] = v0; out[base + t + 256] = v1;
    return;
  }
  red[t] = v0 + v1; __syncthreads();
  for (int o2 = 128; o2; o2 >>= 1) { if (t < o2) red[t] += red[t + o2]; __syncthreads(); }
  float mu = red[0] * (1.0f / DD);
  __syncthreads();
  float d0 = v0 - mu, d1 = v1 - mu;
  red[t] = d0 * d0 + d1 * d1; __syncthreads();
  for (int o2 = 128; o2; o2 >>= 1) { if (t < o2) red[t] += red[t + o2]; __syncthreads(); }
  float rs = rsqrtf(red[0] * (1.0f / DD) + 1e-5f);
  float y0 = d0 * rs * g[t] + beta[t];
  float y1 = d1 * rs * g[t + 256] + beta[t + 256];
  if (do_relu) { y0 = fmaxf(y0, 0.f); y1 = fmaxf(y1, 0.f); }
  out[base + t] = y0; out[base + t + 256] = y1;
}

// ---------------- small-S multi-head attention, block = (b,h,q), 128 thr ----------------
__global__ __launch_bounds__(128) void k_mha(const float* __restrict__ Q, const float* __restrict__ Kk,
                                             const float* __restrict__ V, float* __restrict__ O,
                                             int S, int nh, int dh) {
  __shared__ float sc[160];
  __shared__ float red[128];
  int bid = blockIdx.x;
  int qpos = bid % S;
  int bh = bid / S;
  int h = bh % nh, b = bh / nh;
  int Dm = nh * dh;
  int cpt = dh >> 7;  // dh/128 : 1 or 2
  const float* qr = Q + ((long)(b * S + qpos) * Dm) + h * dh;
  float qreg[2] = {0.f, 0.f};
  for (int j = 0; j < cpt; ++j) qreg[j] = qr[threadIdx.x + j * 128];
  float scale = rsqrtf((float)dh);
  for (int kp = 0; kp < S; ++kp) {
    const float* kr = Kk + ((long)(b * S + kp) * Dm) + h * dh;
    float p = 0.f;
    for (int j = 0; j < cpt; ++j) p += qreg[j] * kr[threadIdx.x + j * 128];
    red[threadIdx.x] = p; __syncthreads();
    for (int o2 = 64; o2; o2 >>= 1) { if (threadIdx.x < o2) red[threadIdx.x] += red[threadIdx.x + o2]; __syncthreads(); }
    if (threadIdx.x == 0) sc[kp] = red[0] * scale;
    __syncthreads();
  }
  if (threadIdx.x == 0) {
    float mx = -1e30f;
    for (int i = 0; i < S; ++i) mx = fmaxf(mx, sc[i]);
    float sm = 0.f;
    for (int i = 0; i < S; ++i) { float e = __expf(sc[i] - mx); sc[i] = e; sm += e; }
    float inv = 1.0f / sm;
    for (int i = 0; i < S; ++i) sc[i] *= inv;
  }
  __syncthreads();
  float o[2] = {0.f, 0.f};
  for (int kp = 0; kp < S; ++kp) {
    const float* vr = V + ((long)(b * S + kp) * Dm) + h * dh;
    float a = sc[kp];
    for (int j = 0; j < cpt; ++j) o[j] += a * vr[threadIdx.x + j * 128];
  }
  float* orow = O + ((long)(b * S + qpos) * Dm) + h * dh;
  for (int j = 0; j < cpt; ++j) orow[threadIdx.x + j * 128] = o[j];
}

// ---------------- gather / absmax / seq build / output ----------------
__global__ void k_gather_rows(const float* __restrict__ src, const int* __restrict__ idx,
                              float* __restrict__ dst) {
  int row = blockIdx.x;
  int s = idx[row];
  for (int d = threadIdx.x; d < DD; d += blockDim.x)
    dst[(long)row * DD + d] = src[(long)s * DD + d];
}

__global__ void k_ring_absmax(const float* __restrict__ t, float* __restrict__ out) {
  long i = (long)blockIdx.x * blockDim.x + threadIdx.x;
  if (i >= (long)RR * DD) return;
  int d = (int)(i % DD);
  int r = (int)(i / DD);
  float best = t[((long)r * RSZ) * DD + d];
  float ba = fabsf(best);
  for (int s = 1; s < RSZ; ++s) {
    float v = t[((long)(r * RSZ + s)) * DD + d];
    float a = fabsf(v);
    if (a > ba) { ba = a; best = v; }
  }
  out[i] = best;
}

__global__ void k_build_seq(const float* __restrict__ h, const float* __restrict__ rv,
                            const float* __restrict__ cls, const float* __restrict__ rt,
                            const float* __restrict__ et, float* __restrict__ seq) {
  long i = (long)blockIdx.x * blockDim.x + threadIdx.x;
  const long total = (long)BB * SEQ * DD;
  if (i >= total) return;
  int d = (int)(i % DD);
  long r = i / DD;
  int pos = (int)(r % SEQ);
  int b = (int)(r / SEQ);
  float v;
  if (pos == 0) v = cls[d];
  else if (pos <= NPER) v = h[((long)b * NPER + (pos - 1)) * DD + d];
  else if (pos == NPER + 1) v = rt[d];
  else if (pos <= NPER + 1 + RPER) v = rv[((long)b * RPER + (pos - NPER - 2)) * DD + d];
  else v = et[d];
  seq[i] = v;
}

__global__ void k_out_write(const float* __restrict__ seq, float* __restrict__ out,
                            long seqn, long total) {
  long i = (long)blockIdx.x * blockDim.x + threadIdx.x;
  if (i >= total) return;
  out[i] = (i < seqn) ? seq[i] : 1.0f;
}

// ================== host orchestration ==================
extern "C" void kernel_launch(void* const* d_in, const int* in_sizes, int n_in,
                              void* d_out, int out_size, void* d_ws, size_t ws_size,
                              hipStream_t stream) {
  (void)in_sizes; (void)n_in; (void)ws_size;
  const float* x          = (const float*)d_in[0];
  const float* edge_attr  = (const float*)d_in[1];
  const int*   edge_index = (const int*)d_in[2];
  const int*   ring_idx   = (const int*)d_in[3];
  const float* x_proj_w   = (const float*)d_in[8];
  const float* x_proj_b   = (const float*)d_in[9];
  const float* e_proj_w   = (const float*)d_in[10];
  const float* e_proj_b   = (const float*)d_in[11];
  const float* gat_wl     = (const float*)d_in[12];
  const float* gat_bl     = (const float*)d_in[13];
  const float* gat_wr     = (const float*)d_in[14];
  const float* gat_br     = (const float*)d_in[15];
  const float* gat_we     = (const float*)d_in[16];
  const float* gat_att    = (const float*)d_in[17];
  const float* gat_bias   = (const float*)d_in[18];
  const float* gat_ln     = (const float*)d_in[19];
  const float* ring_attn_w = (const float*)d_in[20];
  const float* ring_attn_b = (const float*)d_in[21];
  const float* ring_w1    = (const float*)d_in[22];
  const float* ring_b1    = (const float*)d_in[23];
  const float* ring_w2    = (const float*)d_in[24];
  const float* ring_b2    = (const float*)d_in[25];
  const float* ring_ln    = (const float*)d_in[26];
  const float* mol_attn_w = (const float*)d_in[27];
  const float* mol_attn_b = (const float*)d_in[28];
  const float* mol_w1     = (const float*)d_in[29];
  const float* mol_b1     = (const float*)d_in[30];
  const float* mol_w2     = (const float*)d_in[31];
  const float* mol_b2     = (const float*)d_in[32];
  const float* mol_ln     = (const float*)d_in[33];
  const float* cls_tok    = (const float*)d_in[34];
  const float* ring_tok   = (const float*)d_in[35];
  const float* end_tok    = (const float*)d_in[36];

  const int* e_src = edge_index;
  const int* e_dst = edge_index + EE;
  const int ETOT = EE + NN;

  // ---- bump allocator over d_ws ----
  char* ws = (char*)d_ws;
  size_t off = 0;
  auto alloc = [&](size_t bytes) -> void* {
    void* p = ws + off;
    off = (off + bytes + 255) & ~(size_t)255;
    return p;
  };
  const long MAXROWS = (long)BB * SEQ;  // 8896

  unsigned short* wbf_xproj = (unsigned short*)alloc((size_t)XDIM * DD * 2);
  unsigned short* wbf_eproj = (unsigned short*)alloc((size_t)EDIM * DD * 2);
  unsigned short* wbf_wl    = (unsigned short*)alloc((size_t)GATL * DD * DD * 2);
  unsigned short* wbf_wr    = (unsigned short*)alloc((size_t)GATL * DD * DD * 2);
  unsigned short* wbf_we    = (unsigned short*)alloc((size_t)GATL * DD * DD * 2);
  unsigned short* wbf_rattn = (unsigned short*)alloc((size_t)4 * DD * DD * 2);
  unsigned short* wbf_rw1   = (unsigned short*)alloc((size_t)DD * DFF * 2);
  unsigned short* wbf_rw2   = (unsigned short*)alloc((size_t)DFF * DD * 2);
  unsigned short* wbf_mattn = (unsigned short*)alloc((size_t)4 * DD * DD * 2);
  unsigned short* wbf_mw1   = (unsigned short*)alloc((size_t)DD * DFF * 2);
  unsigned short* wbf_mw2   = (unsigned short*)alloc((size_t)DFF * DD * 2);

  unsigned short* x_bf  = (unsigned short*)alloc((size_t)NN * XDIM * 2);
  unsigned short* ea_bf = (unsigned short*)alloc((size_t)EE * EDIM * 2);
  float* h              = (float*)alloc((size_t)NN * DD * 4);
  unsigned short* h_bf  = (unsigned short*)alloc((size_t)NN * DD * 2);
  unsigned short* eall_bf = (unsigned short*)alloc((size_t)ETOT * DD * 2);
  float* BIG            = (float*)alloc((size_t)ETOT * DD * 4);            // e_all f32 / me / FFN hidden
  unsigned short* BIGBF = (unsigned short*)alloc((size_t)MAXROWS * DFF * 2);
  float* bufQ = (float*)alloc((size_t)MAXROWS * DD * 4);
  float* bufK = (float*)alloc((size_t)MAXROWS * DD * 4);
  float* bufV = (float*)alloc((size_t)MAXROWS * DD * 4);
  float* bufO = (float*)alloc((size_t)MAXROWS * DD * 4);
  float* bufY = (float*)alloc((size_t)MAXROWS * DD * 4);
  float* bufT = (float*)alloc((size_t)MAXROWS * DD * 4);
  unsigned short* bfbuf = (unsigned short*)alloc((size_t)MAXROWS * DD * 2);
  float* s_e  = (float*)alloc((size_t)ETOT * 4);
  float* ex_e = (float*)alloc((size_t)ETOT * 4);
  float* z_n  = (float*)alloc((size_t)NN * 4);
  unsigned int* smax_n = (unsigned int*)alloc((size_t)NN * 4);
  float* deg_n = (float*)alloc((size_t)NN * 4);
  float* ringsvec = (float*)alloc((size_t)RR * DD * 4);

  auto cvt = [&](const float* s, unsigned short* d, long n) {
    long nt = (n + 3) / 4;
    k_cvt_bf16<<<(unsigned)((nt + 255) / 256), 256, 0, stream>>>(s, d, n);
  };
  auto fillf = [&](float* p, float v, long n) {
    k_fill_f32<<<(unsigned)((n + 255) / 256), 256, 0, stream>>>(p, v, n);
  };
  auto gemm = [&](const unsigned short* A, const unsigned short* B, const float* bias,
                  float* C, int M, int Nn, int K, int relu) {
    dim3 g((Nn + BN - 1) / BN, (M + BM - 1) / BM);
    if (relu) k_gemm<1><<<g, 256, 0, stream>>>(A, B, bias, C, M, Nn, K);
    else      k_gemm<0><<<g, 256, 0, stream>>>(A, B, bias, C, M, Nn, K);
  };

  // ---- weight conversions ----
  cvt(x_proj_w, wbf_xproj, (long)XDIM * DD);
  cvt(e_proj_w, wbf_eproj, (long)EDIM * DD);
  cvt(gat_wl, wbf_wl, (long)GATL * DD * DD);
  cvt(gat_wr, wbf_wr, (long)GATL * DD * DD);
  cvt(gat_we, wbf_we, (long)GATL * DD * DD);
  cvt(ring_attn_w, wbf_rattn, (long)4 * DD * DD);
  cvt(ring_w1, wbf_rw1, (long)DD * DFF);
  cvt(ring_w2, wbf_rw2, (long)DFF * DD);
  cvt(mol_attn_w, wbf_mattn, (long)4 * DD * DD);
  cvt(mol_w1, wbf_mw1, (long)DD * DFF);
  cvt(mol_w2, wbf_mw2, (long)DFF * DD);

  // ---- input projections ----
  cvt(x, x_bf, (long)NN * XDIM);
  gemm(x_bf, wbf_xproj, x_proj_b, h, NN, DD, XDIM, 0);
  cvt(edge_attr, ea_bf, (long)EE * EDIM);
  gemm(ea_bf, wbf_eproj, e_proj_b, BIG, EE, DD, EDIM, 0);  // e rows [0,E)

  // ---- self-loop attrs: mean incoming e ----
  fillf(BIG + (long)EE * DD, 0.f, (long)NN * DD);
  fillf(deg_n, 0.f, NN);
  k_deg<<<(EE + 255) / 256, 256, 0, stream>>>(e_dst, deg_n);
  k_loop_sum<<<EE, 256, 0, stream>>>(e_dst, BIG);
  k_loop_div<<<NN, 256, 0, stream>>>(deg_n, BIG);
  cvt(BIG, eall_bf, (long)ETOT * DD);
  cvt(h, h_bf, (long)NN * DD);

  // ---- 6 GATv2 layers ----
  for (int l = 0; l < GATL; ++l) {
    gemm(h_bf, wbf_wl + (long)l * DD * DD, gat_bl + l * DD, bufQ, NN, DD, DD, 0);  // xl
    gemm(h_bf, wbf_wr + (long)l * DD * DD, gat_br + l * DD, bufK, NN, DD, DD, 0);  // xr
    gemm(eall_bf, wbf_we + (long)l * DD * DD, nullptr, BIG, ETOT, DD, DD, 0);      // me
    k_fill_u32<<<(NN + 255) / 256, 256, 0, stream>>>(smax_n, 0x007FFFFFu, NN);     // enc(-inf)
    fillf(z_n, 0.f, NN);
    fillf(bufV, 0.f, (long)NN * DD);
    k_edge_scores<<<(ETOT + 7) / 8, 256, 0, stream>>>(bufQ, bufK, BIG, gat_att + l * DD,
                                                      e_src, e_dst, s_e, smax_n, ETOT);
    k_edge_exp<<<(ETOT + 255) / 256, 256, 0, stream>>>(s_e, smax_n, e_dst, ex_e, z_n, ETOT);
    k_edge_agg<<<ETOT, 256, 0, stream>>>(ex_e, z_n, bufQ, e_src, e_dst, bufV);
    if (l < GATL - 1) {
      k_row_ln<<<NN, 256, 0, stream>>>(bufV, gat_bias + l * DD, nullptr,
                                       gat_ln + (long)(l * 2 + 0) * DD,
                                       gat_ln + (long)(l * 2 + 1) * DD, h, 1, 1);
      cvt(h, h_bf, (long)NN * DD);
    } else {
      k_row_ln<<<NN, 256, 0, stream>>>(bufV, gat_bias + l * DD, nullptr,
                                       nullptr, nullptr, h, 0, 0);
    }
  }

  // ---- encoder helper ----
  auto encoder = [&](float* Tio, int Bq, int S, int nh,
                     const unsigned short* attw, const float* attb,
                     const unsigned short* w1, const float* b1,
                     const unsigned short* w2, const float* b2,
                     const float* ln) {
    const long rows = (long)Bq * S;
    const long DD2 = (long)DD * DD;
    cvt(Tio, bfbuf, rows * DD);
    gemm(bfbuf, attw + 0 * DD2, attb + 0 * DD, bufQ, (int)rows, DD, DD, 0);
    gemm(bfbuf, attw + 1 * DD2, attb + 1 * DD, bufK, (int)rows, DD, DD, 0);
    gemm(bfbuf, attw + 2 * DD2, attb + 2 * DD, bufV, (int)rows, DD, DD, 0);
    k_mha<<<Bq * nh * S, 128, 0, stream>>>(bufQ, bufK, bufV, bufO, S, nh, DD / nh);
    cvt(bufO, bfbuf, rows * DD);
    gemm(bfbuf, attw + 3 * DD2, attb + 3 * DD, bufQ, (int)rows, DD, DD, 0);  // attn out
    k_row_ln<<<(unsigned)rows, 256, 0, stream>>>(bufQ, nullptr, Tio, ln + 0 * DD, ln + 1 * DD, bufY, 1, 0);
    cvt(bufY, bfbuf, rows * DD);
    gemm(bfbuf, w1, b1, BIG, (int)rows, DFF, DD, 1);      // relu(y@W1+b1)
    cvt(BIG, BIGBF, rows * DFF);
    gemm(BIGBF, w2, b2, bufQ, (int)rows, DD, DFF, 0);
    k_row_ln<<<(unsigned)rows, 256, 0, stream>>>(bufQ, nullptr, bufY, ln + 2 * DD, ln + 3 * DD, Tio, 1, 0);
  };

  // ---- ring encoder + abs-argmax pooling ----
  k_gather_rows<<<RR * RSZ, 256, 0, stream>>>(h, ring_idx, bufT);
  encoder(bufT, RR, RSZ, 2, wbf_rattn, ring_attn_b, wbf_rw1, ring_b1, wbf_rw2, ring_b2, ring_ln);
  k_ring_absmax<<<((long)RR * DD + 255) / 256, 256, 0, stream>>>(bufT, ringsvec);

  // ---- molecule sequence encoder ----
  {
    const long total = (long)BB * SEQ * DD;
    k_build_seq<<<(unsigned)((total + 255) / 256), 256, 0, stream>>>(h, ringsvec, cls_tok, ring_tok, end_tok, bufT);
  }
  encoder(bufT, BB, SEQ, 4, wbf_mattn, mol_attn_b, wbf_mw1, mol_b1, wbf_mw2, mol_b2, mol_ln);

  // ---- write outputs: seq then all-ones masks ----
  {
    const long seqn = (long)BB * SEQ * DD;
    const long total = (long)out_size;
    k_out_write<<<(unsigned)((total + 255) / 256), 256, 0, stream>>>(bufT, (float*)d_out, seqn, total);
  }
}
